// FullAttention_42923903156804
// MI455X (gfx1250) — matrix operand order repaired and checked by reading, not compiled, not run on hardware
//
#include <hip/hip_runtime.h>
#include <hip/hip_bf16.h>

// Problem constants (from reference)
#define BT      32      // B*T
#define LSEQ    512
#define DMODEL  512
#define NH      8
#define HD      64

typedef __attribute__((ext_vector_type(16))) __bf16 v16bf;
typedef __attribute__((ext_vector_type(8)))  float  v8f;
typedef int v4i __attribute__((vector_size(16)));   // matches async-LDS builtin param

union Frag16 { v16bf v; uint4 q[2]; };

__device__ __forceinline__ __bf16 f2bf(float f) { return (__bf16)f; }

// ---- Fragment loaders (CDNA5 wave32 WMMA 16x16x32 bf16 layouts) ----

// A-matrix 16x32 (MxK), source: f32 row-major, rowp = pointer to this lane's row.
// Lanes 0-15: M=lane, K = {0..7, 16..23}; lanes 16-31: same M, K = {8..15, 24..31}.
__device__ __forceinline__ v16bf loadA_f32(const float* rowp, int kk, int lane) {
  int kb = (lane & 16) ? 8 : 0;
  const float* p = rowp + kk + kb;
  float4 x0 = *(const float4*)(p);
  float4 x1 = *(const float4*)(p + 4);
  float4 x2 = *(const float4*)(p + 16);
  float4 x3 = *(const float4*)(p + 20);
  v16bf a;
  a[0]=f2bf(x0.x); a[1]=f2bf(x0.y); a[2]=f2bf(x0.z); a[3]=f2bf(x0.w);
  a[4]=f2bf(x1.x); a[5]=f2bf(x1.y); a[6]=f2bf(x1.z); a[7]=f2bf(x1.w);
  a[8]=f2bf(x2.x); a[9]=f2bf(x2.y); a[10]=f2bf(x2.z); a[11]=f2bf(x2.w);
  a[12]=f2bf(x3.x); a[13]=f2bf(x3.y); a[14]=f2bf(x3.z); a[15]=f2bf(x3.w);
  return a;
}

// A-matrix 16x32 from bf16 row-major (ld elements per row).
__device__ __forceinline__ v16bf loadA_bf16(const __bf16* base, int ld, int kk, int lane) {
  int row = lane & 15;
  int kb  = (lane & 16) ? 8 : 0;
  const __bf16* p = base + (size_t)row * ld + kk + kb;
  Frag16 f;
  f.q[0] = *(const uint4*)(p);        // K = +0..7
  f.q[1] = *(const uint4*)(p + 16);   // K = +16..23
  return f.v;
}

// B-matrix 32x16 (KxN) where source is stored N-major (row n holds K
// contiguously with leading dim ld): B[k][n] = base[n*ld + k].
// Lanes 0-15: N=lane, K = {0..7, 16..23}; lanes 16-31: N=lane-16, K = {8..15, 24..31}.
__device__ __forceinline__ v16bf loadB_k16(const __bf16* base, int ld, int kk, int lane) {
  int col = lane & 15;
  int kb  = (lane & 16) ? 8 : 0;
  const __bf16* p = base + (size_t)col * ld + kk + kb;
  Frag16 f;
  f.q[0] = *(const uint4*)(p);
  f.q[1] = *(const uint4*)(p + 16);
  return f.v;
}

// ============ Kernel 0: transpose projection weights to bf16 ============
// Wt[((z*NH+h)*HD + n) * DMODEL + k] = W_z[k, h, n]
__global__ __launch_bounds__(256) void prep_w(
    const float* __restrict__ Wq, const float* __restrict__ Wk,
    const float* __restrict__ Wv, __bf16* __restrict__ Wt)
{
  int idx = blockIdx.x * 256 + threadIdx.x;      // 3*512*512 total
  int k  = idx & (DMODEL - 1);
  int n  = (idx >> 9) & (HD - 1);
  int zh = idx >> 15;                            // 0..23
  int z = zh >> 3, h = zh & 7;
  const float* W = (z == 0) ? Wq : (z == 1) ? Wk : Wv;
  Wt[idx] = f2bf(W[(size_t)k * (NH * HD) + h * HD + n]);
}

// ============ Kernel 1: fused Q/K/V projection to bf16 ============
// grid: (row_tiles=128, head=8, which=3), block: 256 threads (8 waves)
__global__ __launch_bounds__(256) void proj_kernel(
    const float* __restrict__ qin, const float* __restrict__ kin,
    const float* __restrict__ vin, const __bf16* __restrict__ Wt,
    const float* __restrict__ w_head,
    __bf16* __restrict__ Qb, __bf16* __restrict__ Kb, __bf16* __restrict__ Vt)
{
  __shared__ __align__(16) __bf16 ldsW[HD * DMODEL];   // W-slice [n][k], 64KB

  const int z = blockIdx.z;
  const float* in = (z == 0) ? qin : (z == 1) ? kin : vin;
  const int head = blockIdx.y;
  const float wscale = (z == 2) ? w_head[head] : 1.0f;   // fold w_head into V

  // Stage transposed W slice: contiguous 64KB global -> LDS copy.
  const int t = threadIdx.x;
  {
    const uint4* g4 = (const uint4*)(Wt + ((size_t)(z * NH + head) * HD) * DMODEL);
    uint4* l4 = (uint4*)ldsW;
#if __has_builtin(__builtin_amdgcn_global_load_async_to_lds_b128)
#pragma unroll
    for (int i = 0; i < 16; ++i) {
      int idx = t + i * 256;
      __builtin_amdgcn_global_load_async_to_lds_b128(
          (__attribute__((address_space(1))) v4i*)(g4 + idx),
          (__attribute__((address_space(3))) v4i*)(l4 + idx), 0, 0);
    }
#if __has_builtin(__builtin_amdgcn_s_wait_asynccnt)
    __builtin_amdgcn_s_wait_asynccnt(0);
#else
    asm volatile("s_wait_asynccnt 0x0" ::: "memory");
#endif
#else
#pragma unroll
    for (int i = 0; i < 16; ++i) {
      int idx = t + i * 256;
      l4[idx] = g4[idx];
    }
#endif
  }
  __syncthreads();

  const int wave = t >> 5, lane = t & 31;
  const int grow = blockIdx.x * 128 + wave * 16;          // row base of wave strip
  const float* arow = in + (size_t)(grow + (lane & 15)) * DMODEL;

  v8f acc[4] = {};
  for (int kk = 0; kk < DMODEL; kk += 32) {
    v16bf a = loadA_f32(arow, kk, lane);
#pragma unroll
    for (int j = 0; j < 4; ++j) {
      v16bf b = loadB_k16(ldsW + j * 16 * DMODEL, DMODEL, kk, lane);
      acc[j] = __builtin_amdgcn_wmma_f32_16x16x32_bf16(
          false, a, false, b, (short)0, acc[j], false, false);
    }
  }

  const int bt    = grow >> 9;
  const int lbase = grow & (LSEQ - 1);
  const int hi    = (lane >> 4) & 1;
  const int nlane = lane & 15;
  if (z == 2) {
    // V stored transposed: (bt, h, e, l)
    __bf16* obase = Vt + ((size_t)(bt * NH + head) * HD) * LSEQ;
#pragma unroll
    for (int j = 0; j < 4; ++j)
#pragma unroll
      for (int r = 0; r < 8; ++r)
        obase[(size_t)(j * 16 + nlane) * LSEQ + (lbase + r + 8 * hi)] =
            f2bf(acc[j][r] * wscale);
  } else {
    // Q/K stored row-major: (bt, h, l, e)
    __bf16* obase = ((z == 0) ? Qb : Kb) + ((size_t)(bt * NH + head) * LSEQ) * HD;
#pragma unroll
    for (int j = 0; j < 4; ++j)
#pragma unroll
      for (int r = 0; r < 8; ++r)
        obase[(size_t)(lbase + r + 8 * hi) * HD + (j * 16 + nlane)] = f2bf(acc[j][r]);
  }
}

// ============ Kernel 2: fused flash attention + head reduction ============
// grid: (L/128 = 4, BT = 32), block 256 (8 waves); wave owns 16 query rows.
// No cross-wave LDS sharing -> no barriers.
__global__ __launch_bounds__(256) void attn_kernel(
    const __bf16* __restrict__ Qb, const __bf16* __restrict__ Kb,
    const __bf16* __restrict__ Vt, const float* __restrict__ mask,
    const float* __restrict__ tau, const float* __restrict__ delta,
    float* __restrict__ outp)
{
  __shared__ __align__(16) __bf16 ldsP[8][16 * 64];    // per-wave P tiles, 16KB

  const int t = threadIdx.x, wave = t >> 5, lane = t & 31;
  const int bt = blockIdx.y;
  const int qstrip = blockIdx.x * 128 + wave * 16;
  const int nlane = lane & 15, hi = (lane >> 4) & 1;
  const float tauv = tau[0], deltav = delta[0];
  const float scale = 0.125f;                           // 1/sqrt(KD=64)

  v8f Ot[4] = {};                                       // head-summed output

  for (int h = 0; h < NH; ++h) {
    const __bf16* Qh = Qb + ((size_t)(bt * NH + h) * LSEQ + qstrip) * HD;
    const __bf16* Kh = Kb + ((size_t)(bt * NH + h) * LSEQ) * HD;
    const __bf16* Vth = Vt + ((size_t)(bt * NH + h) * HD) * LSEQ;   // [e][l]

    v16bf qa0 = loadA_bf16(Qh, HD, 0, lane);
    v16bf qa1 = loadA_bf16(Qh, HD, 32, lane);

    v8f O[4] = {};
    float mrow[8], lrow[8];
#pragma unroll
    for (int r = 0; r < 8; ++r) { mrow[r] = -1e30f; lrow[r] = 0.0f; }

    for (int sb = 0; sb < LSEQ; sb += 64) {
      // Prefetch next chunk of K and V (global_prefetch_b8 path)
      {
        int sp = sb + 64;
        if (sp < LSEQ) {
          __builtin_prefetch(Kh + (size_t)sp * HD + t * 16, 0, 0);
          __builtin_prefetch(Vth + (size_t)(t >> 2) * LSEQ + sp + (t & 3) * 16, 0, 0);
        }
      }

      // Scores: S = Q * K^T  (M=16, N=64 chunk, K=64)
      v8f c[4] = {};
#pragma unroll
      for (int j = 0; j < 4; ++j) {
        v16bf b0 = loadB_k16(Kh + (size_t)(sb + j * 16) * HD, HD, 0, lane);
        c[j] = __builtin_amdgcn_wmma_f32_16x16x32_bf16(
            false, qa0, false, b0, (short)0, c[j], false, false);
        v16bf b1 = loadB_k16(Kh + (size_t)(sb + j * 16) * HD, HD, 32, lane);
        c[j] = __builtin_amdgcn_wmma_f32_16x16x32_bf16(
            false, qa1, false, b1, (short)0, c[j], false, false);
      }

      // De-stationary rescale + scale + multiplicative mask
#pragma unroll
      for (int j = 0; j < 4; ++j) {
        const float* mp = mask + (size_t)qstrip * LSEQ + (sb + j * 16 + nlane);
#pragma unroll
        for (int r = 0; r < 8; ++r) {
          float mval = mp[(size_t)(r + 8 * hi) * LSEQ];
          c[j][r] = (c[j][r] * tauv + deltav) * scale * mval;
        }
      }

      // Online softmax: chunk row max (16-lane xor reductions, wave32)
      float alpha[8], rsum[8];
#pragma unroll
      for (int r = 0; r < 8; ++r) {
        float m0 = fmaxf(fmaxf(c[0][r], c[1][r]), fmaxf(c[2][r], c[3][r]));
#pragma unroll
        for (int xm = 1; xm <= 8; xm <<= 1)
          m0 = fmaxf(m0, __shfl_xor(m0, xm, 32));
        float mnew = fmaxf(mrow[r], m0);
        alpha[r] = __expf(mrow[r] - mnew);
        mrow[r] = mnew;
        rsum[r] = 0.0f;
      }
#pragma unroll
      for (int j = 0; j < 4; ++j)
#pragma unroll
        for (int r = 0; r < 8; ++r) {
          float p = __expf(c[j][r] - mrow[r]);
          c[j][r] = p;
          rsum[r] += p;
        }
#pragma unroll
      for (int r = 0; r < 8; ++r) {
        float s0 = rsum[r];
#pragma unroll
        for (int xm = 1; xm <= 8; xm <<= 1) s0 += __shfl_xor(s0, xm, 32);
        lrow[r] = lrow[r] * alpha[r] + s0;
#pragma unroll
        for (int j = 0; j < 4; ++j) O[j][r] *= alpha[r];
      }

      // Re-fragment P through wave-private LDS (C layout -> A layout)
      __bf16* myP = ldsP[wave];
#pragma unroll
      for (int j = 0; j < 4; ++j)
#pragma unroll
        for (int r = 0; r < 8; ++r)
          myP[(r + 8 * hi) * 64 + j * 16 + nlane] = f2bf(c[j][r]);

      // O += P * V   (M=16, N=64, K=64 chunk); V fragments direct from
      // transposed global V (two 16B runs per lane, 32B apart).
#pragma unroll
      for (int ks = 0; ks < 2; ++ks) {
        v16bf pa = loadA_bf16(myP, 64, ks * 32, lane);
#pragma unroll
        for (int j = 0; j < 4; ++j) {
          v16bf b = loadB_k16(Vth + (size_t)(j * 16) * LSEQ + sb, LSEQ, ks * 32, lane);
          O[j] = __builtin_amdgcn_wmma_f32_16x16x32_bf16(
              false, pa, false, b, (short)0, O[j], false, false);
        }
      }
    }

    // Finalize head: divide by softmax denom, accumulate (w_head already in V)
#pragma unroll
    for (int r = 0; r < 8; ++r) {
      float inv = 1.0f / lrow[r];
#pragma unroll
      for (int j = 0; j < 4; ++j) Ot[j][r] += O[j][r] * inv;
    }
  }

  // Write output (bt, l, e) f32
  float* ob = outp + ((size_t)bt * LSEQ + qstrip) * HD;
#pragma unroll
  for (int j = 0; j < 4; ++j)
#pragma unroll
    for (int r = 0; r < 8; ++r)
      ob[(size_t)(r + 8 * hi) * HD + (j * 16 + nlane)] = Ot[j][r];
}

extern "C" void kernel_launch(void* const* d_in, const int* in_sizes, int n_in,
                              void* d_out, int out_size, void* d_ws, size_t ws_size,
                              hipStream_t stream) {
  (void)in_sizes; (void)n_in; (void)out_size; (void)ws_size;
  const float* queries = (const float*)d_in[0];
  const float* keys    = (const float*)d_in[1];
  const float* values  = (const float*)d_in[2];
  const float* mask    = (const float*)d_in[3];
  const float* Wq      = (const float*)d_in[4];
  const float* Wk      = (const float*)d_in[5];
  const float* Wv      = (const float*)d_in[6];
  const float* w_head  = (const float*)d_in[7];
  const float* tau     = (const float*)d_in[8];
  const float* delta   = (const float*)d_in[9];
  float* out = (float*)d_out;

  const size_t nproj = (size_t)BT * LSEQ * NH * HD;   // 8,388,608 elems per tensor
  __bf16* Qb = (__bf16*)d_ws;
  __bf16* Kb = Qb + nproj;
  __bf16* Vt = Kb + nproj;
  __bf16* Wt = Vt + nproj;                            // 3*512*512 bf16

  prep_w<<<dim3(3 * DMODEL * DMODEL / 256), 256, 0, stream>>>(Wq, Wk, Wv, Wt);

  dim3 pgrid(128, NH, 3);
  proj_kernel<<<pgrid, 256, 0, stream>>>(queries, keys, values, Wt, w_head,
                                         Qb, Kb, Vt);
  dim3 agrid(LSEQ / 128, BT);
  attn_kernel<<<agrid, 256, 0, stream>>>(Qb, Kb, Vt, mask, tau, delta, out);
}
